// KANBasisLinear_40252433498123
// MI455X (gfx1250) — compile-verified, matrix operand order checked by reading, not run on hardware
//
#include <hip/hip_runtime.h>
#include <hip/hip_bf16.h>

typedef __attribute__((ext_vector_type(16))) _Float16 v16h;
typedef __attribute__((ext_vector_type(8)))  _Float16 v8h;
typedef __attribute__((ext_vector_type(8)))  float    v8f;

// Problem constants (match reference)
static constexpr int BATCH = 16384;
static constexpr int IN_F  = 1024;
static constexpr int OUT_F = 1024;
static constexpr int KAUG  = 10;            // 9 hat-basis knots + 1 silu slot
static constexpr int KTOT  = IN_F * KAUG;   // 10240 augmented inner dim

// Tiling
static constexpr int BM   = 128;
static constexpr int BN   = 256;
static constexpr int BKP  = 160;            // K panel = lcm(32,10): slot index j is compile-time
static constexpr int SUBK = 32;             // WMMA K step
static constexpr int NSUB = BKP / SUBK;     // 5 sub-steps per panel
static constexpr int IPP  = BKP / KAUG;     // 16 input features per panel
static constexpr int LDA  = 168;            // As stride (halfs): 336B = 21*16B -> conflict-free b128
static constexpr int LDB  = 40;             // Bs stride (halfs): 80B = 5*16B  -> conflict-free b128
static constexpr int NTHREADS = 512;        // 16 wave32 waves, 4(M) x 4(N)

// ---------------------------------------------------------------------------
// Kernel 1: pack spline_weight[O][I][9] + base_weight[O][I] into f16
// Wp[n][k'] with k' = i*10 + j  (j<9: spline knot j, j==9: base weight).
// ---------------------------------------------------------------------------
__global__ __launch_bounds__(256) void pack_weights(
    const float* __restrict__ sw, const float* __restrict__ bw,
    _Float16* __restrict__ wp) {
  int idx = blockIdx.x * 256 + threadIdx.x;          // 0 .. OUT_F*KTOT-1
  int n = idx / KTOT;
  int k = idx - n * KTOT;
  int i = k / KAUG;
  int j = k - i * KAUG;
  float v = (j < 9) ? sw[(n * IN_F + i) * 9 + j] : bw[n * IN_F + i];
  wp[idx] = (_Float16)v;
}

// ---------------------------------------------------------------------------
// Kernel 2: fused KAN GEMM:  out = Aug(x) @ Wp + bias
// Aug(x)[b][i*10+j] = j<9 ? relu(1 - |2*clip(x,-2,2)+4 - j|) : silu(x)
// ---------------------------------------------------------------------------
__global__ __launch_bounds__(NTHREADS) void kan_wmma(
    const float* __restrict__ x, const _Float16* __restrict__ wp,
    const float* __restrict__ bias, float* __restrict__ out) {
  __shared__ _Float16 As[BM * LDA];   // 43008 B : one 160-K activation panel
  __shared__ _Float16 Bs[BN * LDB];   // 20480 B : one 32-K weight chunk

  const int tid  = threadIdx.x;
  const int lane = tid & 31;
  const int wave = tid >> 5;        // 0..15
  const int wm   = wave & 3;        // wave M position -> 32 rows
  const int wn   = wave >> 2;       // wave N position -> 64 cols
  const int mBlk = blockIdx.x * BM;
  const int nBlk = blockIdx.y * BN;

  const int l15 = lane & 15;
  const int khi = (lane >> 4) * 8;  // fragment k-base / C-row base for upper half-wave

  // As fill: thread -> (row, group of 4 input features) ; 40 halfs = 5 x b128
  const int arow = tid >> 2;            // 0..127
  const int agrp = tid & 3;             // 0..3
  const float* xrow = x + (size_t)(mBlk + arow) * IN_F + agrp * 4;
  _Float16* adst = &As[arow * LDA + agrp * 40];

  // Bs fill: thread -> (col row, 16-half chunk)
  const int brow  = tid >> 1;           // 0..255
  const int bhalf = (tid & 1) * 16;     // 0 or 16
  const _Float16* wrow = wp + (size_t)(nBlk + brow) * KTOT + bhalf;
  _Float16* bdst = &Bs[brow * LDB + bhalf];

  v8f acc[2][4];
#pragma unroll
  for (int mf = 0; mf < 2; ++mf)
#pragma unroll
    for (int nf = 0; nf < 4; ++nf)
      acc[mf][nf] = (v8f)(0.0f);

  for (int p = 0; p < KTOT / BKP; ++p) {        // 64 panels
    // ---- stage activation panel As[128][160]: j is compile-time, no branches
    {
      union { _Float16 h[40]; v8h v[5]; } t;
#pragma unroll
      for (int q = 0; q < 4; ++q) {
        float xv = xrow[p * IPP + q];
        float xc = fminf(fmaxf(xv, -2.0f), 2.0f);
        float u  = __builtin_fmaf(xc, 2.0f, 4.0f);       // in [0,8]
#pragma unroll
        for (int j = 0; j < 9; ++j) {
          float d  = u - (float)j;
          float tv = fmaxf(__builtin_fmaf(fabsf(d), -1.0f, 1.0f), 0.0f);
          t.h[q * 10 + j] = (_Float16)tv;
        }
        float s = xv / (1.0f + __expf(-xv));             // silu
        t.h[q * 10 + 9] = (_Float16)s;
      }
#pragma unroll
      for (int q = 0; q < 5; ++q) *((v8h*)adst + q) = t.v[q];
    }

#pragma unroll 1
    for (int s = 0; s < NSUB; ++s) {
      // ---- stage 32-K weight chunk (L2-resident packed f16) ----
      {
        const _Float16* wr = wrow + p * BKP + s * SUBK;
        v8h w0 = *(const v8h*)(wr);
        v8h w1 = *(const v8h*)(wr + 8);
        *(v8h*)bdst       = w0;
        *(v8h*)(bdst + 8) = w1;
        __builtin_prefetch(wr + SUBK, 0, 1);
      }
      __syncthreads();

      // ---- fragments: ISA 16-bit A/B layout = two 16B chunks per lane ----
      v16h afrag[2], bfrag[4];
#pragma unroll
      for (int mf = 0; mf < 2; ++mf) {
        const _Float16* pa = &As[(wm * 32 + mf * 16 + l15) * LDA + s * SUBK + khi];
        v8h c0 = *(const v8h*)pa;
        v8h c1 = *(const v8h*)(pa + 16);
        afrag[mf] = __builtin_shufflevector(c0, c1, 0, 1, 2, 3, 4, 5, 6, 7,
                                            8, 9, 10, 11, 12, 13, 14, 15);
      }
#pragma unroll
      for (int nf = 0; nf < 4; ++nf) {
        const _Float16* pb = &Bs[(wn * 64 + nf * 16 + l15) * LDB + khi];
        v8h c0 = *(const v8h*)pb;
        v8h c1 = *(const v8h*)(pb + 16);
        bfrag[nf] = __builtin_shufflevector(c0, c1, 0, 1, 2, 3, 4, 5, 6, 7,
                                            8, 9, 10, 11, 12, 13, 14, 15);
      }

      // ---- 8 WMMAs: wave tile 32x64, f32 accumulate ----
#pragma unroll
      for (int mf = 0; mf < 2; ++mf)
#pragma unroll
        for (int nf = 0; nf < 4; ++nf)
          acc[mf][nf] = __builtin_amdgcn_wmma_f32_16x16x32_f16(
              false, afrag[mf], false, bfrag[nf], (short)0, acc[mf][nf],
              false, false);

      __syncthreads();
    }
  }

  // ---- epilogue: C layout lane(0-15)=N, vgpr e -> M = base + (lane>=16?8:0) + e
#pragma unroll
  for (int nf = 0; nf < 4; ++nf) {
    int n = nBlk + wn * 64 + nf * 16 + l15;
    float bv = bias[n];
#pragma unroll
    for (int mf = 0; mf < 2; ++mf) {
      int mBase = mBlk + wm * 32 + mf * 16 + khi;
#pragma unroll
      for (int e = 0; e < 8; ++e)
        out[(size_t)(mBase + e) * OUT_F + n] = acc[mf][nf][e] + bv;
    }
  }
}

// ---------------------------------------------------------------------------
extern "C" void kernel_launch(void* const* d_in, const int* in_sizes, int n_in,
                              void* d_out, int out_size, void* d_ws, size_t ws_size,
                              hipStream_t stream) {
  const float* x    = (const float*)d_in[0];   // [B, I]
  const float* sw   = (const float*)d_in[1];   // [O, I, 9]
  const float* bw   = (const float*)d_in[2];   // [O, I]
  const float* bias = (const float*)d_in[3];   // [O]
  float* out = (float*)d_out;                  // [B, O]

  _Float16* wp = (_Float16*)d_ws;              // needs OUT_F*KTOT*2 = 20 MiB

  pack_weights<<<(OUT_F * KTOT) / 256, 256, 0, stream>>>(sw, bw, wp);

  dim3 grid(BATCH / BM, OUT_F / BN);           // 128 x 4 workgroups
  kan_wmma<<<grid, NTHREADS, 0, stream>>>(x, wp, bias, out);
}
